// MultiheadAttention_38208029065557
// MI455X (gfx1250) — matrix operand order repaired and checked by reading, not compiled, not run on hardware
//
#include <hip/hip_runtime.h>
#include <cstdint>

typedef __bf16 bf16;
typedef __attribute__((ext_vector_type(16))) __bf16 v16bf;
typedef __attribute__((ext_vector_type(8)))  float  v8f;
typedef __attribute__((ext_vector_type(4)))  float  vf4;
typedef __attribute__((ext_vector_type(4)))  int    i32x4;

#define LQ   4096
#define NB   2
#define EMB  512
#define NH   8
#define DH   64
#define MTOT (LQ * NB)        // 8192 rows in projection space

// ---------------------------------------------------------------------------
// Async global->LDS copy (16B per lane), CDNA5 GLOBAL_LOAD_ASYNC_TO_LDS_B128.
// Builtin signature (verified via hipcc diagnostic): (AS1 int4*, AS3 int4*,
// imm offset, imm cpol). Tracked by ASYNCcnt.
// ---------------------------------------------------------------------------
#if __has_builtin(__builtin_amdgcn_global_load_async_to_lds_b128)
#define HAVE_ASYNC 1
#else
#define HAVE_ASYNC 0
#endif

__device__ __forceinline__ void cp16(void* dst_lds, const void* src_global) {
#if HAVE_ASYNC
  void* s = const_cast<void*>(src_global);
  __builtin_amdgcn_global_load_async_to_lds_b128(
      (__attribute__((address_space(1))) i32x4*)s,
      (__attribute__((address_space(3))) i32x4*)dst_lds, 0, 0);
#else
  *(uint4*)dst_lds = *(const uint4*)src_global;
#endif
}
__device__ __forceinline__ void wait_async() {
#if HAVE_ASYNC
#if __has_builtin(__builtin_amdgcn_s_wait_asynccnt)
  __builtin_amdgcn_s_wait_asynccnt(0);
#else
  asm volatile("s_wait_asynccnt 0x0" ::: "memory");
#endif
#endif
}

// ---------------------------------------------------------------------------
// WMMA fragment helpers (CDNA5 ISA §7.12.2, wave32)
// A 16x32 bf16: lane half h, row = lane%16; elems j<8 -> K = base+j,
//               j>=8 -> K = base+16+(j-8), base = kbase + h*8
// B 32x16 bf16 (LDS holds B^T as [ncol][K]): same K order as A
// ---------------------------------------------------------------------------
__device__ __forceinline__ v16bf frag_a(const bf16* p) {
  v16bf r;
#pragma unroll
  for (int j = 0; j < 8; ++j) r[j] = p[j];
#pragma unroll
  for (int j = 0; j < 8; ++j) r[j + 8] = p[j + 16];
  return r;
}
__device__ __forceinline__ v16bf frag_b(const bf16* p) {
  v16bf r;
#pragma unroll
  for (int j = 0; j < 8; ++j) r[j] = p[j];
#pragma unroll
  for (int j = 0; j < 8; ++j) r[j + 8] = p[j + 16];
  return r;
}
__device__ __forceinline__ v8f wmma_bf16(v16bf a, v16bf b, v8f c) {
  return __builtin_amdgcn_wmma_f32_16x16x32_bf16(false, a, false, b, (short)0, c,
                                                 false, false);
}

// ---------------------------------------------------------------------------
// Weight prep: plain f32 -> bf16 convert, layout preserved ([e][k] row-major).
// Row-major W is already the B^T layout the WMMA B-fragment wants for A·W^T.
// ---------------------------------------------------------------------------
__global__ __launch_bounds__(256) void conv_w(const float* __restrict__ W,
                                              bf16* __restrict__ Wb) {
  int idx = blockIdx.x * 256 + threadIdx.x;   // 0 .. 512*512-1, fully coalesced
  Wb[idx] = (bf16)W[idx];
}

// ---------------------------------------------------------------------------
// GEMM: C[m, c] = (sum_k A[m,k] * W[c,k] + bias[c]) * scale
// A: [8192,512] (f32 or bf16), Wb: [512,512] bf16 row-major.
// Block tile 64(M) x 128(N), k-tiles of 64, double-buffered async staging.
// 256 threads = 8 waves in 4(M) x 2(N) grid; wave strip = 16 x 64 (4 tiles).
// MODE 0: bf16 out [8192,512]
// MODE 1: f32 out with (l,n) row permutation (final output projection)
// MODE 2: bf16 out transposed per head: VpT[((n*8+h)*64+d)*4096 + l]
// ---------------------------------------------------------------------------
template <bool A_IS_F32, int MODE>
__global__ __launch_bounds__(256) void gemm512(const void* __restrict__ Aptr,
                                               const bf16* __restrict__ Wb,
                                               const float* __restrict__ bias,
                                               void* __restrict__ Out,
                                               float scale) {
  __shared__ bf16 Alds[2][64][80];
  __shared__ bf16 Blds[2][128][80];
  const int tid  = threadIdx.x;
  const int lane = tid & 31, wv = tid >> 5;
  const int hi = lane >> 4, nc = lane & 15;
  const int m0 = blockIdx.x * 64;
  const int c0 = blockIdx.y * 128;
  const int Msub = (wv & 3) * 16;
  const int cb   = (wv >> 2) * 64;

  auto stageA = [&](int k0, int buf) {
    const int r = tid >> 2, seg = (tid & 3) * 16;
    if (A_IS_F32) {
      const float* A = (const float*)Aptr + (size_t)(m0 + r) * 512 + k0 + seg;
      const float4* Av = (const float4*)A;
      float4 f0 = Av[0], f1 = Av[1], f2 = Av[2], f3 = Av[3];
      union { bf16 b[16]; uint4 u[2]; } t;
      t.b[0]  = (bf16)f0.x; t.b[1]  = (bf16)f0.y; t.b[2]  = (bf16)f0.z; t.b[3]  = (bf16)f0.w;
      t.b[4]  = (bf16)f1.x; t.b[5]  = (bf16)f1.y; t.b[6]  = (bf16)f1.z; t.b[7]  = (bf16)f1.w;
      t.b[8]  = (bf16)f2.x; t.b[9]  = (bf16)f2.y; t.b[10] = (bf16)f2.z; t.b[11] = (bf16)f2.w;
      t.b[12] = (bf16)f3.x; t.b[13] = (bf16)f3.y; t.b[14] = (bf16)f3.z; t.b[15] = (bf16)f3.w;
      uint4* d = (uint4*)&Alds[buf][r][seg];
      d[0] = t.u[0]; d[1] = t.u[1];
    } else {
      const bf16* A = (const bf16*)Aptr + (size_t)(m0 + r) * 512 + k0 + seg;
      cp16(&Alds[buf][r][seg], A);
      cp16(&Alds[buf][r][seg + 8], A + 8);
    }
  };
  auto stageB = [&](int k0, int buf) {
    const int r = tid >> 1, seg = (tid & 1) * 32;   // 128 rows x 64 k
    const bf16* B = Wb + (size_t)(c0 + r) * 512 + k0 + seg;
#pragma unroll
    for (int i = 0; i < 4; ++i) cp16(&Blds[buf][r][seg + i * 8], B + i * 8);
  };

  v8f acc[4] = {};
  stageA(0, 0);
  stageB(0, 0);
  for (int it = 0; it < 8; ++it) {
    wait_async();
    __syncthreads();
    if (it + 1 < 8) {
      stageA((it + 1) * 64, (it + 1) & 1);
      stageB((it + 1) * 64, (it + 1) & 1);
    }
    const int b = it & 1;
#pragma unroll
    for (int kk = 0; kk < 64; kk += 32) {
      v16bf a = frag_a(&Alds[b][Msub + nc][kk + hi * 8]);
#pragma unroll
      for (int t = 0; t < 4; ++t) {
        v16bf bb = frag_b(&Blds[b][cb + t * 16 + nc][kk + hi * 8]);
        acc[t] = wmma_bf16(a, bb, acc[t]);
      }
    }
  }
  // ---- epilogue ----
#pragma unroll
  for (int t = 0; t < 4; ++t) {
    const int col = c0 + cb + t * 16 + nc;
    const float bv = bias[col];
#pragma unroll
    for (int r = 0; r < 8; ++r) {
      const int m = m0 + Msub + hi * 8 + r;
      float v = (acc[t][r] + bv) * scale;
      if (MODE == 0) {
        ((bf16*)Out)[(size_t)m * 512 + col] = (bf16)v;
      } else if (MODE == 1) {
        const int orow = (m & 4095) * NB + (m >> 12);  // (l, n) permutation
        ((float*)Out)[(size_t)orow * 512 + col] = v;
      } else {
        const int hh = col >> 6, d = col & 63;
        const int nn = m >> 12, l = m & 4095;
        ((bf16*)Out)[(((size_t)nn * NH + hh) * DH + d) * LQ + l] = (bf16)v;
      }
    }
  }
}

// ---------------------------------------------------------------------------
// Pass 1: raw scaled scores S = Q K^T (scale folded into Q) + online softmax
// stats (rowmax, rowsumexp). Block = 64 q-rows of one (n,h). K tiles of 128,
// async double-buffered. Stats are accumulated PER LANE (no cross-lane traffic
// in the hot loop) and merged across the 16 lanes / 2 column-group waves once
// at the end using the online-softmax combine.
// ---------------------------------------------------------------------------
__global__ __launch_bounds__(256) void attn_scores(const bf16* __restrict__ Qp,
                                                   const bf16* __restrict__ Kp,
                                                   float* __restrict__ attn,
                                                   float2* __restrict__ stats) {
  __shared__ bf16 Qlds[64][80];
  __shared__ bf16 Klds[2][128][80];
  __shared__ float smx[2][64];
  __shared__ float ssm[2][64];
  const int tid  = threadIdx.x;
  const int lane = tid & 31, wv = tid >> 5;
  const int hi = lane >> 4, nc = lane & 15;
  const int qb = blockIdx.x, h = blockIdx.y, n = blockIdx.z;
  const int Msub = (wv & 3) * 16;
  const int cb   = (wv >> 2) * 64;
  const size_t qrow0 = (size_t)n * LQ + qb * 64;

  auto stageK = [&](int kt, int buf) {
    const int r = tid >> 1, seg = (tid & 1) * 32;
    const bf16* src = Kp + ((size_t)n * LQ + kt + r) * 512 + h * 64 + seg;
#pragma unroll
    for (int i = 0; i < 4; ++i) cp16(&Klds[buf][r][seg + i * 8], src + i * 8);
  };

  // stage Q tile (already scaled by 1/8 at projection); kick K tile 0
  {
    const int r = tid >> 2, seg = (tid & 3) * 16;
    const bf16* src = Qp + (qrow0 + r) * 512 + h * 64 + seg;
    const uint4* sv = (const uint4*)src;
    uint4* d = (uint4*)&Qlds[r][seg];
    d[0] = sv[0]; d[1] = sv[1];
  }
  stageK(0, 0);
  __syncthreads();
  const v16bf a0 = frag_a(&Qlds[Msub + nc][0 + hi * 8]);
  const v16bf a1 = frag_a(&Qlds[Msub + nc][32 + hi * 8]);

  float mrun[8], srun[8];
#pragma unroll
  for (int r = 0; r < 8; ++r) { mrun[r] = -3.0e38f; srun[r] = 0.0f; }

  float* attn_base = attn + ((size_t)(n * NH + h) * LQ + qb * 64) * (size_t)LQ;

  for (int it = 0; it < LQ / 128; ++it) {
    const int kt = it * 128;
    wait_async();
    __syncthreads();
    if (it + 1 < LQ / 128) stageK(kt + 128, (it + 1) & 1);
    const int b = it & 1;
    v8f c[4] = {};
#pragma unroll
    for (int t = 0; t < 4; ++t)
      c[t] = wmma_bf16(a0, frag_b(&Klds[b][cb + t * 16 + nc][0 + hi * 8]), c[t]);
#pragma unroll
    for (int t = 0; t < 4; ++t)
      c[t] = wmma_bf16(a1, frag_b(&Klds[b][cb + t * 16 + nc][32 + hi * 8]), c[t]);

    // write raw scores (non-temporal); per-lane online (max, sumexp) update
#pragma unroll
    for (int r = 0; r < 8; ++r) {
      const int mrow = Msub + hi * 8 + r;
      float v0 = c[0][r], v1 = c[1][r], v2 = c[2][r], v3 = c[3][r];
      float* o = attn_base + (size_t)mrow * LQ + kt + cb + nc;
      __builtin_nontemporal_store(v0, o);
      __builtin_nontemporal_store(v1, o + 16);
      __builtin_nontemporal_store(v2, o + 32);
      __builtin_nontemporal_store(v3, o + 48);
      float tm = fmaxf(fmaxf(v0, v1), fmaxf(v2, v3));
      float mn = fmaxf(mrun[r], tm);
      float ps = __expf(v0 - mn) + __expf(v1 - mn) + __expf(v2 - mn) + __expf(v3 - mn);
      srun[r] = srun[r] * __expf(mrun[r] - mn) + ps;
      mrun[r] = mn;
    }
  }
  // merge the 16 lanes of each half-wave (online-softmax pairwise combine)
#pragma unroll
  for (int r = 0; r < 8; ++r) {
    float m = mrun[r], s = srun[r];
#pragma unroll
    for (int d = 1; d < 16; d <<= 1) {
      float mo = __shfl_xor(m, d, 32);
      float so = __shfl_xor(s, d, 32);
      float mn = fmaxf(m, mo);
      s = s * __expf(m - mn) + so * __expf(mo - mn);
      m = mn;
    }
    mrun[r] = m; srun[r] = s;
  }
  // combine the two column-group waves sharing each row
  if (nc == 0) {
#pragma unroll
    for (int r = 0; r < 8; ++r) {
      smx[wv >> 2][Msub + hi * 8 + r] = mrun[r];
      ssm[wv >> 2][Msub + hi * 8 + r] = srun[r];
    }
  }
  __syncthreads();
  if (tid < 64) {
    float m0 = smx[0][tid], m1 = smx[1][tid];
    float m = fmaxf(m0, m1);
    float s = ssm[0][tid] * __expf(m0 - m) + ssm[1][tid] * __expf(m1 - m);
    stats[(size_t)(n * NH + h) * LQ + qb * 64 + tid] = make_float2(m, s);
  }
}

// ---------------------------------------------------------------------------
// Pass 2: normalize weights (final attn_weights to d_out) and AV GEMM.
// Block = 64 q-rows of one (n,h); k-tiles of 64, double-buffered:
// V tiles async from the pre-transposed VpT; weight tiles flow through VALU.
// Wave grid 4(M) x 2(cols of 32); out = 64 x 64 (Dh).
// ---------------------------------------------------------------------------
__global__ __launch_bounds__(256) void attn_av(float* __restrict__ attn,
                                               const float2* __restrict__ stats,
                                               const bf16* __restrict__ VpT,
                                               bf16* __restrict__ AO) {
  __shared__ bf16 Wlds[2][64][80];
  __shared__ bf16 Vlds[2][64][80];
  const int tid  = threadIdx.x;
  const int lane = tid & 31, wv = tid >> 5;
  const int hi = lane >> 4, nc = lane & 15;
  const int qb = blockIdx.x, h = blockIdx.y, n = blockIdx.z;
  const int Msub = (wv & 3) * 16;
  const int cb   = (wv >> 2) * 32;
  float* attn_base = attn + ((size_t)(n * NH + h) * LQ + qb * 64) * (size_t)LQ;
  const bf16* Vb = VpT + ((size_t)(n * NH + h) * DH) * LQ;

  const int nr   = tid >> 2;         // row this thread normalizes
  const int nseg = (tid & 3) * 16;   // 16-col segment
  const float2 st = stats[(size_t)(n * NH + h) * LQ + qb * 64 + nr];
  const float mrow = st.x;
  const float inv  = 1.0f / st.y;

  auto normalize = [&](int kt, int buf) {
    float* p = attn_base + (size_t)nr * LQ + kt + nseg;
    vf4 f[4];
#pragma unroll
    for (int i = 0; i < 4; ++i) f[i] = __builtin_nontemporal_load((const vf4*)p + i);
    union { bf16 b[16]; uint4 u[2]; } t;
#pragma unroll
    for (int i = 0; i < 4; ++i)
#pragma unroll
      for (int j = 0; j < 4; ++j) {
        float w = __expf(f[i][j] - mrow) * inv;
        f[i][j] = w;
        t.b[i * 4 + j] = (bf16)w;
      }
#pragma unroll
    for (int i = 0; i < 4; ++i) __builtin_nontemporal_store(f[i], (vf4*)p + i);
    uint4* d = (uint4*)&Wlds[buf][nr][nseg];
    d[0] = t.u[0]; d[1] = t.u[1];
  };
  auto stageV = [&](int kt, int buf) {
    const int r = tid >> 2, seg = (tid & 3) * 16;   // r = d (0..63)
    const bf16* src = Vb + (size_t)r * LQ + kt + seg;
    cp16(&Vlds[buf][r][seg], src);
    cp16(&Vlds[buf][r][seg + 8], src + 8);
  };

  v8f acc[2] = {};
  stageV(0, 0);
  normalize(0, 0);
  for (int it = 0; it < LQ / 64; ++it) {
    wait_async();
    __syncthreads();
    if (it + 1 < LQ / 64) {
      stageV((it + 1) * 64, (it + 1) & 1);
      normalize((it + 1) * 64, (it + 1) & 1);
    }
    const int b = it & 1;
#pragma unroll
    for (int kk = 0; kk < 64; kk += 32) {
      v16bf a = frag_a(&Wlds[b][Msub + nc][kk + hi * 8]);
#pragma unroll
      for (int t = 0; t < 2; ++t) {
        v16bf bb = frag_b(&Vlds[b][cb + t * 16 + nc][kk + hi * 8]);
        acc[t] = wmma_bf16(a, bb, acc[t]);
      }
    }
  }
  // store head output (bf16) into AO [8192, 512]
#pragma unroll
  for (int t = 0; t < 2; ++t) {
    const int col = h * 64 + cb + t * 16 + nc;
#pragma unroll
    for (int r = 0; r < 8; ++r) {
      const size_t m = (size_t)n * LQ + qb * 64 + Msub + hi * 8 + r;
      AO[m * 512 + col] = (bf16)acc[t][r];
    }
  }
}

// ---------------------------------------------------------------------------
// Launcher
// ---------------------------------------------------------------------------
extern "C" void kernel_launch(void* const* d_in, const int* in_sizes, int n_in,
                              void* d_out, int out_size, void* d_ws, size_t ws_size,
                              hipStream_t stream) {
  const float* q  = (const float*)d_in[0];
  const float* k  = (const float*)d_in[1];
  const float* v  = (const float*)d_in[2];
  const float* Wq = (const float*)d_in[3];
  const float* bq = (const float*)d_in[4];
  const float* Wk = (const float*)d_in[5];
  const float* bk = (const float*)d_in[6];
  const float* Wv = (const float*)d_in[7];
  const float* bv = (const float*)d_in[8];
  const float* Wo = (const float*)d_in[9];
  const float* bo = (const float*)d_in[10];

  float* out   = (float*)d_out;
  float* attnw = out + (size_t)LQ * NB * EMB;   // attn_weights region

  char* wsp = (char*)d_ws;
  bf16* Wqb = (bf16*)wsp; wsp += (size_t)512 * 512 * sizeof(bf16);
  bf16* Wkb = (bf16*)wsp; wsp += (size_t)512 * 512 * sizeof(bf16);
  bf16* Wvb = (bf16*)wsp; wsp += (size_t)512 * 512 * sizeof(bf16);
  bf16* Wob = (bf16*)wsp; wsp += (size_t)512 * 512 * sizeof(bf16);
  bf16* Qp  = (bf16*)wsp; wsp += (size_t)MTOT * 512 * sizeof(bf16);
  bf16* Kp  = (bf16*)wsp; wsp += (size_t)MTOT * 512 * sizeof(bf16);
  bf16* VpT = (bf16*)wsp; wsp += (size_t)MTOT * 512 * sizeof(bf16);
  bf16* AO  = (bf16*)wsp; wsp += (size_t)MTOT * 512 * sizeof(bf16);
  float2* stats = (float2*)wsp;

  conv_w<<<1024, 256, 0, stream>>>(Wq, Wqb);
  conv_w<<<1024, 256, 0, stream>>>(Wk, Wkb);
  conv_w<<<1024, 256, 0, stream>>>(Wv, Wvb);
  conv_w<<<1024, 256, 0, stream>>>(Wo, Wob);

  dim3 gg(MTOT / 64, EMB / 128);
  gemm512<true, 0><<<gg, 256, 0, stream>>>(q, Wqb, bq, Qp, 0.125f);  // 1/sqrt(Dh) folded
  gemm512<true, 0><<<gg, 256, 0, stream>>>(k, Wkb, bk, Kp, 1.0f);
  gemm512<true, 2><<<gg, 256, 0, stream>>>(v, Wvb, bv, VpT, 1.0f);   // per-head transpose

  dim3 ga(LQ / 64, NH, NB);
  attn_scores<<<ga, 256, 0, stream>>>(Qp, Kp, attnw, stats);
  attn_av<<<ga, 256, 0, stream>>>(attnw, stats, VpT, AO);

  gemm512<false, 1><<<gg, 256, 0, stream>>>(AO, Wob, bo, out, 1.0f);
}
